// ConvolutionLayer_55722905698549
// MI455X (gfx1250) — matrix operand order repaired and checked by reading, not compiled, not run on hardware
//
#include <hip/hip_runtime.h>

// ---------------------------------------------------------------------------
// TBCNN convolution layer for MI455X (gfx1250, wave32, WMMA).
// B=32, N=512, C=16, D=256, O=256.
//
// Dominant work: (B*N, 3D) x (3D, O) GEMM = 6.44 GFLOP -> matrix-core bound.
// bf16 WMMA (16x16x32), f32 accumulation. B slices double-buffered in LDS via
// GLOBAL_LOAD_ASYNC_TO_LDS_B128 (ASYNCcnt). Wave tile = 64M x 32N (4x2 WMMA
// tiles): each B-fragment feeds 4 WMMAs -> 4x less LDS read bandwidth than a
// 1x8 tile, same A/B global traffic.
// ---------------------------------------------------------------------------

typedef __attribute__((ext_vector_type(16))) __bf16    v16bf;
typedef __attribute__((ext_vector_type(8)))  float     v8f;
typedef __attribute__((ext_vector_type(4)))  float     f32x4;
typedef __attribute__((ext_vector_type(4)))  unsigned  u32x4;

#define BB   32
#define NN   512
#define CC   16
#define DD   256
#define OO   256
#define MTOT (BB * NN)   // 16384
#define KTOT (3 * DD)    // 768
#define KSTEPS (KTOT / 32)

// round-to-nearest-even f32 -> bf16, packed pair into one dword
__device__ __forceinline__ unsigned f2bf_pack(float a, float b) {
  union { float f; unsigned u; } x, y;
  x.f = a; y.f = b;
  unsigned ra = (x.u + 0x7FFFu + ((x.u >> 16) & 1u)) >> 16;
  unsigned rb = (y.u + 0x7FFFu + ((y.u >> 16) & 1u)) >> 16;
  return ra | (rb << 16);
}

// ---------------------------------------------------------------------------
// Kernel 1: pack [w_t; w_r; w_l] (each (D,O) f32 row-major) into bf16 B-matrix
// of shape (KTOT, OO). Order t,r,l matches the coef stack (c_t, c_r, c_l).
// ---------------------------------------------------------------------------
__global__ __launch_bounds__(256) void tbcnn_wpack(
    const float* __restrict__ wt, const float* __restrict__ wr,
    const float* __restrict__ wl, unsigned short* __restrict__ Bw) {
  int u = blockIdx.x * 256 + threadIdx.x;    // dword index, < 98304
  int e = u * 2;                             // bf16 element index
  int sec = e >> 16;
  int off = e & 65535;
  const float* src = (sec == 0) ? wt : (sec == 1) ? wr : wl;
  ((unsigned*)Bw)[u] = f2bf_pack(src[off], src[off + 1]);
}

// ---------------------------------------------------------------------------
// Kernel 2: gather + coefficient aggregation -> bf16 A-matrix (MTOT, KTOT).
// One wave per node; lane owns 8 of the 256 d-values.
// Row layout: [0,256)=agg_t (parent), [256,512)=agg_r, [512,768)=agg_l.
// ---------------------------------------------------------------------------
__global__ __launch_bounds__(256) void tbcnn_prep(
    const float* __restrict__ nodes, const int* __restrict__ children,
    unsigned short* __restrict__ Abuf) {
  const int lane = threadIdx.x & 31;
  const int node = blockIdx.x * 8 + (threadIdx.x >> 5);   // < 16384
  const int b = node >> 9;
  const int n = node & 511;

  const float* base = nodes + (size_t)b * (NN * DD);
  const f32x4* pp = (const f32x4*)(base + (size_t)n * DD + lane * 8);
  f32x4 p0 = pp[0], p1 = pp[1];
  float pe[8] = { p0.x, p0.y, p0.z, p0.w, p1.x, p1.y, p1.z, p1.w };

  const int* ch = children + (size_t)node * CC;
  int cidx[CC];
  int ns = 0;
#pragma unroll
  for (int j = 0; j < CC; ++j) {
    cidx[j] = ch[j];
    ns += (cidx[j] != 0);
  }
  const bool  single = (ns == 1);
  const float denom  = single ? 1.0f : (float)(ns - 1);  // ns==0 -> -1 (masked)
  const float inv    = 1.0f / denom;

  float re[8], le[8];
#pragma unroll
  for (int k = 0; k < 8; ++k) { re[k] = 0.0f; le[k] = 0.0f; }

#pragma unroll
  for (int j = 0; j < CC; ++j) {
    int cj = cidx[j];
    if (cj != 0) {
      float cr = single ? (j == 0 ? 0.5f : 0.0f) : (float)j * inv;
      float cl = 1.0f - cr;
      const f32x4* ep = (const f32x4*)(base + (size_t)cj * DD + lane * 8);
      f32x4 e0 = ep[0], e1 = ep[1];
      float ce[8] = { e0.x, e0.y, e0.z, e0.w, e1.x, e1.y, e1.z, e1.w };
#pragma unroll
      for (int k = 0; k < 8; ++k) {
        re[k] = fmaf(cr, ce[k], re[k]);
        le[k] = fmaf(cl, ce[k], le[k]);
      }
    }
  }

  unsigned short* row = Abuf + (size_t)node * KTOT;
  u32x4 qt, qr, ql;
#pragma unroll
  for (int i = 0; i < 4; ++i) {
    qt[i] = f2bf_pack(pe[2 * i], pe[2 * i + 1]);
    qr[i] = f2bf_pack(re[2 * i], re[2 * i + 1]);
    ql[i] = f2bf_pack(le[2 * i], le[2 * i + 1]);
  }
  *(u32x4*)(row +       lane * 8) = qt;
  *(u32x4*)(row + 256 + lane * 8) = qr;
  *(u32x4*)(row + 512 + lane * 8) = ql;
}

// ---------------------------------------------------------------------------
// Kernel 3: bf16 WMMA GEMM (M=16384, K=768, N=256) + bias + leaky_relu.
// Block: 256 thr (8 waves). Block tile: 128(M) x 128(N).
// Wave tile: 64(M) x 32(N) = 4x2 WMMA tiles (2 waves in M x 4 waves in N).
// K in 32-deep slices, double-buffered in LDS via async-to-LDS copies.
// ---------------------------------------------------------------------------
#define LDS_STRIDE 136   // ushorts per B row: 128 + 8 pad (16B aligned, bank spread)

__global__ __launch_bounds__(256) void tbcnn_gemm(
    const unsigned short* __restrict__ A, const unsigned short* __restrict__ Bw,
    const float* __restrict__ conv, float* __restrict__ out) {
  __shared__ unsigned short ldsB[2][32 * LDS_STRIDE];   // 2 x 8704 B

  const int lane   = threadIdx.x & 31;
  const int wave   = threadIdx.x >> 5;
  const int half   = lane >> 4;
  const int mrow   = lane & 15;
  const int mGroup = wave & 1;        // 2 wave-groups in M
  const int nGroup = wave >> 1;       // 4 wave-groups in N
  const int nBaseB = (blockIdx.x & 1) * 128;              // block N offset
  const int mBase  = (blockIdx.x >> 1) * 128 + mGroup * 64;
  const int nOff   = nGroup * 32;                         // within block slice

  v8f acc[4][2] = {};

  // A base for this lane (ISA 16-bit A 16x32 layout: lane = M%16); the 4
  // M-tiles sit at immediate offsets mt*16*KTOT*2 = mt*24576 bytes.
  const unsigned* aBase = (const unsigned*)(A + (size_t)(mBase + mrow) * KTOT);

  // async staging: thread -> (row lr, 32B chunk lc) of the 32x128 slice
  const int lr = threadIdx.x >> 3;   // 0..31
  const int lc = threadIdx.x & 7;    // 0..7
  const unsigned long long gsrc =
      (unsigned long long)(uintptr_t)Bw +
      (unsigned long long)((lr * OO + nBaseB + lc * 32) * 2);
  // generic->LDS: low 32 bits of a flat LDS pointer are the LDS byte offset
  const unsigned la0 = (unsigned)(uintptr_t)&ldsB[0][lr * LDS_STRIDE + lc * 16];
  const unsigned la1 = (unsigned)(uintptr_t)&ldsB[1][lr * LDS_STRIDE + lc * 16];

  auto stage = [&](int slice, int buf) {
    unsigned long long ga = gsrc + (unsigned long long)slice * (32u * OO * 2u);
    unsigned la = buf ? la1 : la0;
    // IOFFSET applies to both global and LDS addresses for async ops,
    // so offset:16 covers the second 16-byte chunk of this thread's 32 bytes.
    asm volatile(
        "global_load_async_to_lds_b128 %0, %1, off\n\t"
        "global_load_async_to_lds_b128 %0, %1, off offset:16"
        :: "v"(la), "v"(ga) : "memory");
  };

  stage(0, 0);   // prologue: slice 0 -> buffer 0

  for (int k = 0; k < KSTEPS; ++k) {
    const int p = k & 1;
    asm volatile("s_wait_asynccnt 0x0" ::: "memory");  // my LDS writes landed
    __syncthreads();                                   // everyone's landed

    if (k + 1 < KSTEPS) stage(k + 1, 1 - p);           // overlap next fetch

    // A fragments: 4 M-tiles, each two 16B loads (one clause, imm offsets).
    // VGPR i<4 -> K = 32k + 8*half + 2i ; i>=4 -> +16.
    union { v16bf v; u32x4 q[2]; } af[4];
    const u32x4* ap = (const u32x4*)(aBase + k * 16 + half * 4);
#pragma unroll
    for (int mt = 0; mt < 4; ++mt) {
      const u32x4* apm = (const u32x4*)((const char*)ap + mt * (16 * KTOT * 2));
      af[mt].q[0] = apm[0];
      af[mt].q[1] = apm[2];
    }

    // 2 B-fragments (lane = N col, 16 bf16 = K in A's table order), each reused by the
    // 4 WMMAs of its column strip (independent acc/A -> back-to-back issue).
    const unsigned short* bs = &ldsB[p][nOff + mrow];
    union { v16bf v; unsigned short s[16]; } bfr[2];
#pragma unroll
    for (int nt = 0; nt < 2; ++nt) {
#pragma unroll
      for (int s = 0; s < 16; ++s) {
        const int kk = (s < 8) ? (8 * half + s) : (16 + 8 * half + (s - 8));
        bfr[nt].s[s] = bs[kk * LDS_STRIDE + nt * 16];
      }
    }
#pragma unroll
    for (int nt = 0; nt < 2; ++nt) {
#pragma unroll
      for (int mt = 0; mt < 4; ++mt) {
        acc[mt][nt] = __builtin_amdgcn_wmma_f32_16x16x32_bf16(
            false, af[mt].v, false, bfr[nt].v, (short)0, acc[mt][nt],
            false, false);
      }
    }
  }

  // Epilogue. C/D layout: VGPR r -> M = r + 8*half, N = lane&15.
#pragma unroll
  for (int nt = 0; nt < 2; ++nt) {
    const int col  = nBaseB + nOff + nt * 16 + mrow;
    const float bias = conv[col];
#pragma unroll
    for (int mt = 0; mt < 4; ++mt) {
#pragma unroll
      for (int r = 0; r < 8; ++r) {
        const int row = mBase + mt * 16 + r + 8 * half;
        float v = acc[mt][nt][r] + bias;
        v = (v > 0.0f) ? v : 0.01f * v;
        out[(size_t)row * OO + col] = v;
      }
    }
  }
}

// ---------------------------------------------------------------------------
extern "C" void kernel_launch(void* const* d_in, const int* in_sizes, int n_in,
                              void* d_out, int out_size, void* d_ws, size_t ws_size,
                              hipStream_t stream) {
  (void)in_sizes; (void)n_in; (void)out_size; (void)ws_size;
  const float* nodes    = (const float*)d_in[0];
  const float* w_t      = (const float*)d_in[1];
  const float* w_l      = (const float*)d_in[2];
  const float* w_r      = (const float*)d_in[3];
  const float* conv     = (const float*)d_in[4];
  const int*   children = (const int*)d_in[5];

  // workspace layout: bf16 B (768x256) = 393216 B, then bf16 A (16384x768)
  unsigned short* Bw   = (unsigned short*)d_ws;
  unsigned short* Abuf = (unsigned short*)((char*)d_ws + 393216);

  tbcnn_wpack<<<384, 256, 0, stream>>>(w_t, w_r, w_l, Bw);
  tbcnn_prep <<<MTOT / 8, 256, 0, stream>>>(nodes, children, Abuf);
  tbcnn_gemm <<<256, 256, 0, stream>>>(Abuf, Bw, conv, (float*)d_out);
}
